// LinearBlock_82746839924988
// MI455X (gfx1250) — matrix operand order repaired and checked by reading, not compiled, not run on hardware
//
#include <hip/hip_runtime.h>
#include <hip/hip_bf16.h>

typedef _Float16 v16h __attribute__((ext_vector_type(16)));
typedef _Float16 v8h  __attribute__((ext_vector_type(8)));
typedef float    v8f  __attribute__((ext_vector_type(8)));

#define TB  2048
#define CC  1024
#define HH  16
#define DD  64
#define FFD 4096

// ---- CDNA5 async copy: global -> LDS, 16 bytes per lane, tracked by ASYNCcnt ----
__device__ __forceinline__ void async_cp16(void* lds_ptr, const void* gptr) {
  unsigned lds_off = (unsigned)(uintptr_t)lds_ptr;   // low 32 bits of generic ptr = LDS byte address
  asm volatile("global_load_async_to_lds_b128 %0, %1, off"
               :: "v"(lds_off), "v"(gptr) : "memory");
}
__device__ __forceinline__ void wait_async_le4() {
  asm volatile("s_wait_asynccnt 0x4" ::: "memory");
}
__device__ __forceinline__ void wait_async_0() {
  asm volatile("s_wait_asynccnt 0x0" ::: "memory");
}

// ---------------- transpose + convert: in f32 [K][N] -> out f16 [N][K] ----------------
__global__ __launch_bounds__(256)
void transpose_cvt(const float* __restrict__ in, _Float16* __restrict__ out, int K, int N) {
  __shared__ float tile[64][65];
  const int tid = threadIdx.x;
  const int kb = blockIdx.y * 64, nb = blockIdx.x * 64;
#pragma unroll
  for (int p = 0; p < 16; ++p) {
    int idx = p * 256 + tid;
    int r = idx >> 6, c = idx & 63;
    tile[r][c] = in[(size_t)(kb + r) * N + (nb + c)];
  }
  __syncthreads();
#pragma unroll
  for (int p = 0; p < 16; ++p) {
    int idx = p * 256 + tid;
    int r = idx >> 6, c = idx & 63;
    out[(size_t)(nb + r) * K + (kb + c)] = (_Float16)tile[c][r];
  }
}

// ---------------- LayerNorm: one block (256 thr) per row of C=1024 ----------------
__global__ __launch_bounds__(256)
void ln_kernel(const float* __restrict__ x, const float* __restrict__ g,
               const float* __restrict__ b, _Float16* __restrict__ out) {
  __shared__ float s1[256], s2[256];
  const int row = blockIdx.x, tid = threadIdx.x;
  const float* xr = x + (size_t)row * CC;
  float v[4]; float sum = 0.f, sq = 0.f;
#pragma unroll
  for (int i = 0; i < 4; ++i) { v[i] = xr[tid * 4 + i]; sum += v[i]; sq += v[i] * v[i]; }
  s1[tid] = sum; s2[tid] = sq; __syncthreads();
  for (int off = 128; off > 0; off >>= 1) {
    if (tid < off) { s1[tid] += s1[tid + off]; s2[tid] += s2[tid + off]; }
    __syncthreads();
  }
  const float mu = s1[0] * (1.0f / CC);
  const float var = s2[0] * (1.0f / CC) - mu * mu;
  const float rs = rsqrtf(var + 1e-5f);
#pragma unroll
  for (int i = 0; i < 4; ++i) {
    const int c = tid * 4 + i;
    out[(size_t)row * CC + c] = (_Float16)((v[i] - mu) * rs * g[c] + b[c]);
  }
}

// ---------------- WMMA GEMM: C(MxN,f32) = A(MxK,f16) * Bt(NxK,f16)^T + bias ----------------
// Block tile 128x128, BK=32, double-buffered async global->LDS staging.
// 8 waves in 4x2; each wave: 32x64 = 2x4 WMMA tiles (8 WMMAs per K-step).
#define LDA 40   // padded f16 stride for A tile rows (128 x 32)
#define LDB 40   // padded f16 stride for Bt tile rows (128 x 32)
__global__ __launch_bounds__(256)
void gemm_wmma(const _Float16* __restrict__ A, const _Float16* __restrict__ Bt,
               const float* __restrict__ bias, const float* __restrict__ res,
               float* __restrict__ outF, _Float16* __restrict__ outH,
               int M, int N, int K, int act) {
  __shared__ __align__(16) _Float16 As[2][128 * LDA];
  __shared__ __align__(16) _Float16 Bs[2][128 * LDB];

  const int tid  = threadIdx.x;
  const int lane = tid & 31;
  const int wid  = tid >> 5;
  const int wr   = wid >> 1;       // wave row 0..3  (32 rows each)
  const int wc   = wid & 1;        // wave col 0..1  (64 cols each)
  const int hf   = lane >> 4;      // half-wave 0/1
  const int lm   = lane & 15;
  const int rowBase = blockIdx.y * 128;
  const int colBase = blockIdx.x * 128;

  v8f acc[2][4];
#pragma unroll
  for (int i = 0; i < 2; ++i)
#pragma unroll
    for (int j = 0; j < 4; ++j)
#pragma unroll
      for (int r = 0; r < 8; ++r) acc[i][j][r] = 0.0f;

  // staging map: each thread moves 4x16B per tile (A rows ar, ar+64; Bt rows ar, ar+64)
  const int ar = tid >> 2;          // 0..63
  const int ac = (tid & 3) * 8;     // 0,8,16,24  (f16 elements)

  const _Float16* Ag = A  + (size_t)(rowBase + ar) * K + ac;
  const _Float16* Bg = Bt + (size_t)(colBase + ar) * K + ac;

  const int nIter = K >> 5;

  // prologue: stage tile 0 into buffer 0
  {
    async_cp16(&As[0][ar * LDA + ac],        Ag);
    async_cp16(&As[0][(ar + 64) * LDA + ac], Ag + (size_t)64 * K);
    async_cp16(&Bs[0][ar * LDB + ac],        Bg);
    async_cp16(&Bs[0][(ar + 64) * LDB + ac], Bg + (size_t)64 * K);
  }

  for (int it = 0; it < nIter; ++it) {
    const int cur = it & 1;
    __syncthreads();                       // all waves done reading buf[cur^1]
    if (it + 1 < nIter) {
      const int kt = (it + 1) << 5;
      async_cp16(&As[cur ^ 1][ar * LDA + ac],        Ag + kt);
      async_cp16(&As[cur ^ 1][(ar + 64) * LDA + ac], Ag + (size_t)64 * K + kt);
      async_cp16(&Bs[cur ^ 1][ar * LDB + ac],        Bg + kt);
      async_cp16(&Bs[cur ^ 1][(ar + 64) * LDB + ac], Bg + (size_t)64 * K + kt);
      wait_async_le4();                    // tile `it` complete; tile `it+1` in flight
    } else {
      wait_async_0();
    }
    __syncthreads();                       // buf[cur] visible to all waves

    v16h af[2], bf[4];
#pragma unroll
    for (int tm = 0; tm < 2; ++tm) {
      const _Float16* p = &As[cur][(wr * 32 + tm * 16 + lm) * LDA + hf * 8];
      v8h lo = *(const v8h*)p;
      v8h hi = *(const v8h*)(p + 16);
#pragma unroll
      for (int i = 0; i < 8; ++i) { af[tm][i] = lo[i]; af[tm][i + 8] = hi[i]; }
    }
#pragma unroll
    for (int tn = 0; tn < 4; ++tn) {
      const _Float16* p = &Bs[cur][(wc * 64 + tn * 16 + lm) * LDB + hf * 8];
      v8h lo = *(const v8h*)p;
      v8h hi = *(const v8h*)(p + 16);
#pragma unroll
      for (int i = 0; i < 8; ++i) { bf[tn][i] = lo[i]; bf[tn][i + 8] = hi[i]; }
    }
#pragma unroll
    for (int tm = 0; tm < 2; ++tm)
#pragma unroll
      for (int tn = 0; tn < 4; ++tn)
        acc[tm][tn] = __builtin_amdgcn_wmma_f32_16x16x32_f16(
            false, af[tm], false, bf[tn], (short)0, acc[tm][tn], false, false);
  }

#pragma unroll
  for (int tm = 0; tm < 2; ++tm)
#pragma unroll
    for (int tn = 0; tn < 4; ++tn)
#pragma unroll
      for (int r = 0; r < 8; ++r) {
        const int row = rowBase + wr * 32 + tm * 16 + hf * 8 + r;
        const int col = colBase + wc * 64 + tn * 16 + lm;
        float vv = acc[tm][tn][r] + bias[col];
        if (act == 1) vv = 0.5f * vv * (1.0f + erff(vv * 0.70710678118654752f));
        if (res)  vv += res[(size_t)row * N + col];
        if (outF) outF[(size_t)row * N + col] = vv;
        if (outH) outH[(size_t)row * N + col] = (_Float16)vv;
      }
}

// ---------------- Causal linear attention scan: one block per head ----------------
__global__ __launch_bounds__(256)
void attn_kernel(const float* __restrict__ qkv, _Float16* __restrict__ o16) {
  __shared__ float q_s[64], k_s[64], v_s[64], Kc[64], part[4][64];
  const int h = blockIdx.x, tid = threadIdx.x;
  const int e = tid & 63, g = tid >> 6, dbase = g * 16;
  float st[16];
#pragma unroll
  for (int i = 0; i < 16; ++i) st[i] = 0.0f;
  if (tid < 64) Kc[tid] = 0.0f;
  __syncthreads();

  for (int t = 0; t < TB; ++t) {
    const float* base = qkv + (size_t)t * (3 * CC) + h * DD;
    if (tid < 64) {
      float qv = base[tid];
      q_s[tid] = qv > 0.f ? qv + 1.0f : expf(qv);
    } else if (tid < 128) {
      int i = tid - 64;
      float kv = base[CC + i];
      k_s[i] = kv > 0.f ? kv + 1.0f : expf(kv);
    } else if (tid < 192) {
      int i = tid - 128;
      v_s[i] = base[2 * CC + i];
    }
    __syncthreads();
    if (tid < 64) Kc[tid] += k_s[tid];
    const float ve = v_s[e];
    float p = 0.0f;
#pragma unroll
    for (int i = 0; i < 16; ++i) {
      float s = st[i] + k_s[dbase + i] * ve;
      st[i] = s;
      p += q_s[dbase + i] * s;
    }
    part[g][e] = p;
    __syncthreads();
    if (tid < 64) {
      float num = part[0][tid] + part[1][tid] + part[2][tid] + part[3][tid];
      float den = 0.0f;
      for (int d = 0; d < 64; ++d) den += q_s[d] * Kc[d];
      o16[(size_t)t * CC + h * DD + tid] = (_Float16)(num / (den + 1e-6f));
    }
    __syncthreads();
  }
}

extern "C" void kernel_launch(void* const* d_in, const int* in_sizes, int n_in,
                              void* d_out, int out_size, void* d_ws, size_t ws_size,
                              hipStream_t stream) {
  const float* x     = (const float*)d_in[0];
  const float* qkv_w = (const float*)d_in[1];
  const float* qkv_b = (const float*)d_in[2];
  const float* out_w = (const float*)d_in[3];
  const float* out_b = (const float*)d_in[4];
  const float* ln1_g = (const float*)d_in[5];
  const float* ln1_b = (const float*)d_in[6];
  const float* ln2_g = (const float*)d_in[7];
  const float* ln2_b = (const float*)d_in[8];
  const float* w1    = (const float*)d_in[9];
  const float* b1    = (const float*)d_in[10];
  const float* w2    = (const float*)d_in[11];
  const float* b2    = (const float*)d_in[12];
  float* out = (float*)d_out;

  char* ws = (char*)d_ws;
  size_t off = 0;
  auto alloc = [&](size_t bytes) -> void* {
    void* p = ws + off;
    off += (bytes + 255) & ~(size_t)255;
    return p;
  };
  _Float16* h16   = (_Float16*)alloc((size_t)TB * CC * 2);        // LN output (reused for LN2)
  _Float16* wq16  = (_Float16*)alloc((size_t)CC * 3 * CC * 2);    // transposed: (3C, C)
  _Float16* wo16  = (_Float16*)alloc((size_t)CC * CC * 2);        // transposed: (C, C)
  _Float16* w1_16 = (_Float16*)alloc((size_t)CC * FFD * 2);       // transposed: (FF, C)
  _Float16* w2_16 = (_Float16*)alloc((size_t)FFD * CC * 2);       // transposed: (C, FF)
  float*    qkvf  = (float*)   alloc((size_t)TB * 3 * CC * 4);    // later reused as act16
  _Float16* o16   = (_Float16*)alloc((size_t)TB * CC * 2);
  float*    x1    = (float*)   alloc((size_t)TB * CC * 4);
  _Float16* act16 = (_Float16*)qkvf;                               // reuse (16MB <= 24MB)

  // weight transpose+convert: f32 [K][N] -> f16 [N][K]
  transpose_cvt<<<dim3(3 * CC / 64, CC / 64), 256, 0, stream>>>(qkv_w, wq16, CC, 3 * CC);
  transpose_cvt<<<dim3(CC / 64, CC / 64), 256, 0, stream>>>(out_w, wo16, CC, CC);
  transpose_cvt<<<dim3(FFD / 64, CC / 64), 256, 0, stream>>>(w1, w1_16, CC, FFD);
  transpose_cvt<<<dim3(CC / 64, FFD / 64), 256, 0, stream>>>(w2, w2_16, FFD, CC);

  // LN1
  ln_kernel<<<TB, 256, 0, stream>>>(x, ln1_g, ln1_b, h16);

  // qkv = h @ qkv_w + qkv_b  (2048 x 3072, K=1024)
  gemm_wmma<<<dim3(3 * CC / 128, TB / 128), 256, 0, stream>>>(
      h16, wq16, qkv_b, nullptr, qkvf, nullptr, TB, 3 * CC, CC, 0);

  // causal linear attention -> o16 (f16)
  attn_kernel<<<HH, 256, 0, stream>>>(qkvf, o16);

  // x1 = x + o @ out_w + out_b  (2048 x 1024, K=1024)
  gemm_wmma<<<dim3(CC / 128, TB / 128), 256, 0, stream>>>(
      o16, wo16, out_b, x, x1, nullptr, TB, CC, CC, 0);

  // LN2
  ln_kernel<<<TB, 256, 0, stream>>>(x1, ln2_g, ln2_b, h16);

  // act = gelu(h2 @ w1 + b1)  (2048 x 4096, K=1024), f16 out
  gemm_wmma<<<dim3(FFD / 128, TB / 128), 256, 0, stream>>>(
      h16, w1_16, b1, nullptr, nullptr, act16, TB, FFD, CC, 1);

  // out = x1 + act @ w2 + b2  (2048 x 1024, K=4096)
  gemm_wmma<<<dim3(CC / 128, TB / 128), 256, 0, stream>>>(
      act16, w2_16, b2, x1, out, nullptr, TB, CC, FFD, 0);

  (void)in_sizes; (void)n_in; (void)out_size; (void)ws_size;
}
